// LinearMPC_15006615733275
// MI455X (gfx1250) — matrix operand order repaired and checked by reading, not compiled, not run on hardware
//
#include <hip/hip_runtime.h>

#define NXC    8
#define NC     64
#define MC     512
#define BATCHC 2048
#define ITERS  100
#define STEPC  0.01f

typedef __attribute__((ext_vector_type(16))) _Float16 v16h;
typedef __attribute__((ext_vector_type(8)))  _Float16 v8h;
typedef __attribute__((ext_vector_type(8)))  float    v8f;

// ---------------------------------------------------------------------------
// Prep: H (f32, [M][M], row m, col k) -> f16 copy. B-operand of the WMMA is
// B[k][n] = H[m0+n][k], i.e. a contiguous row of this buffer per lane.
// ---------------------------------------------------------------------------
__global__ __launch_bounds__(256) void mpc_prep_h(const float* __restrict__ H,
                                                  _Float16* __restrict__ Hh) {
  int i = blockIdx.x * blockDim.x + threadIdx.x;   // grid covers M*M
  Hh[i] = (_Float16)H[i];
}

// ---------------------------------------------------------------------------
// Prep: f[b, k*8+i] = -2 * sum_j Phi[k,i,j] * (Q @ (xref[b,k]-xref[b,0]))[j]
// One thread per (b, k): 128 MACs. ~131k threads.
// ---------------------------------------------------------------------------
__global__ __launch_bounds__(256) void mpc_compute_f(const float* __restrict__ xref,
                                                     const float* __restrict__ Phi,
                                                     const float* __restrict__ Q,
                                                     float* __restrict__ F) {
  int idx = blockIdx.x * blockDim.x + threadIdx.x;   // b*NC + k
  int b = idx / NC;
  int k = idx % NC;
  const float* xr = xref + (size_t)b * (NC + 1) * NXC;

  float dx[NXC], qd[NXC];
#pragma unroll
  for (int l = 0; l < NXC; ++l)
    dx[l] = xr[k * NXC + l] - xr[l];
#pragma unroll
  for (int j = 0; j < NXC; ++j) {
    float s = 0.f;
#pragma unroll
    for (int l = 0; l < NXC; ++l) s += Q[j * NXC + l] * dx[l];
    qd[j] = s;
  }
  const float* Pk = Phi + (size_t)k * NXC * NXC;
#pragma unroll
  for (int i = 0; i < NXC; ++i) {
    float s = 0.f;
#pragma unroll
    for (int j = 0; j < NXC; ++j) s += Pk[i * NXC + j] * qd[j];
    F[(size_t)b * MC + k * NXC + i] = -2.0f * s;
  }
}

// ---------------------------------------------------------------------------
// Prep: u0 = 0 (both f32 master and f16 shadow).
// ---------------------------------------------------------------------------
__global__ __launch_bounds__(256) void mpc_zero_u(float* __restrict__ Uf,
                                                  _Float16* __restrict__ Uh) {
  int i = blockIdx.x * blockDim.x + threadIdx.x;   // grid covers BATCH*M
  Uf[i] = 0.0f;
  Uh[i] = (_Float16)0.0f;
}

// ---------------------------------------------------------------------------
// One PGD iteration:  u_new = clip(u - step*(H u + f), -1, 1)
// grad[b,m] = sum_k u[b,k] * H[m,k]  via v_wmma_f32_16x16x32_f16.
// Wave computes a 16(batch) x 64(m) strip: A fragment reused across 4 tiles.
// Block = 8 waves -> 128 batch rows x 64 m cols. Grid = (16, 8).
// ---------------------------------------------------------------------------
__global__ __launch_bounds__(256) void mpc_pgd_iter(
    const _Float16* __restrict__ Hh,   // [M][M] f16 (row m, col k)
    const _Float16* __restrict__ Uh,   // [B][M] f16 current u
    const float*    __restrict__ Uf,   // [B][M] f32 current u
    const float*    __restrict__ F,    // [B][M] f32 linear term
    float*          __restrict__ UfN,  // [B][M] f32 next u
    _Float16*       __restrict__ UhN)  // [B][M] f16 next u
{
  const int lane = threadIdx.x & 31;
  const int wave = threadIdx.x >> 5;                 // 0..7
  const int b0 = (blockIdx.x * 8 + wave) * 16;       // batch tile base
  const int m0 = blockIdx.y * 64;                    // m strip base
  const int lhalf = lane >> 4;                       // 0 | 1
  const int lidx  = lane & 15;

  // A-operand (16-bit 16x32): lane<16 -> K in {0..15}; lane>=16 -> {16..31}
  const _Float16* Abase = Uh + (size_t)(b0 + lidx) * MC + lhalf * 16;
  // B-operand (32x16): lane n holds H row (m0 + t*16 + n), K block +0 (lanes 0-15) / +16 (16-31)
  const _Float16* Bbase = Hh + (size_t)(m0 + lidx) * MC + lhalf * 16;

  v8f acc0 = {}, acc1 = {}, acc2 = {}, acc3 = {};

#pragma unroll 4
  for (int k0 = 0; k0 < MC; k0 += 32) {
    v8h alo = *(const v8h*)(Abase + k0);
    v8h ahi = *(const v8h*)(Abase + k0 + 8);
    v16h a = __builtin_shufflevector(alo, ahi, 0, 1, 2, 3, 4, 5, 6, 7,
                                     8, 9, 10, 11, 12, 13, 14, 15);
    v16h bt0 = *(const v16h*)(Bbase + 0 * 16 * MC + k0);
    v16h bt1 = *(const v16h*)(Bbase + 1 * 16 * MC + k0);
    v16h bt2 = *(const v16h*)(Bbase + 2 * 16 * MC + k0);
    v16h bt3 = *(const v16h*)(Bbase + 3 * 16 * MC + k0);
    acc0 = __builtin_amdgcn_wmma_f32_16x16x32_f16(false, a, false, bt0, (short)0, acc0, false, false);
    acc1 = __builtin_amdgcn_wmma_f32_16x16x32_f16(false, a, false, bt1, (short)0, acc1, false, false);
    acc2 = __builtin_amdgcn_wmma_f32_16x16x32_f16(false, a, false, bt2, (short)0, acc2, false, false);
    acc3 = __builtin_amdgcn_wmma_f32_16x16x32_f16(false, a, false, bt3, (short)0, acc3, false, false);
  }

  // C/D layout: VGPR r, lane -> batch = b0 + r + 8*(lane>=16), m = m0 + t*16 + (lane&15)
  const int radd = lhalf * 8;
  v8f accs[4] = {acc0, acc1, acc2, acc3};
#pragma unroll
  for (int t = 0; t < 4; ++t) {
#pragma unroll
    for (int r = 0; r < 8; ++r) {
      const size_t idx = (size_t)(b0 + r + radd) * MC + (m0 + t * 16 + lidx);
      float g  = accs[t][r] + F[idx];
      float un = Uf[idx] - STEPC * g;
      un = fminf(fmaxf(un, -1.0f), 1.0f);
      UfN[idx] = un;
      UhN[idx] = (_Float16)un;
    }
  }
}

// ---------------------------------------------------------------------------
// Host: prep + 100 ping-pong iterations. Final (iter 100, even count) lands in
// buffer 0 = d_out.
// ---------------------------------------------------------------------------
extern "C" void kernel_launch(void* const* d_in, const int* in_sizes, int n_in,
                              void* d_out, int out_size, void* d_ws, size_t ws_size,
                              hipStream_t stream) {
  (void)in_sizes; (void)n_in; (void)out_size; (void)ws_size;
  // setup_inputs order: x0, xref, H, Phi, Q  (x0 is unused by the reference)
  const float* xref = (const float*)d_in[1];
  const float* H    = (const float*)d_in[2];
  const float* Phi  = (const float*)d_in[3];
  const float* Q    = (const float*)d_in[4];

  char* ws = (char*)d_ws;
  _Float16* Hh  = (_Float16*)ws;  ws += (size_t)MC * MC * 2;          // 512 KB
  _Float16* Uh0 = (_Float16*)ws;  ws += (size_t)BATCHC * MC * 2;      // 2 MB
  _Float16* Uh1 = (_Float16*)ws;  ws += (size_t)BATCHC * MC * 2;      // 2 MB
  float*    Uf1 = (float*)ws;     ws += (size_t)BATCHC * MC * 4;      // 4 MB
  float*    F   = (float*)ws;     ws += (size_t)BATCHC * MC * 4;      // 4 MB
  float*    Uf0 = (float*)d_out;                                      // master u ping buffer

  mpc_prep_h   <<<(MC * MC) / 256,     256, 0, stream>>>(H, Hh);
  mpc_compute_f<<<(BATCHC * NC) / 256, 256, 0, stream>>>(xref, Phi, Q, F);
  mpc_zero_u   <<<(BATCHC * MC) / 256, 256, 0, stream>>>(Uf0, Uh0);

  float*    ufs[2] = {Uf0, Uf1};
  _Float16* uhs[2] = {Uh0, Uh1};
  for (int it = 0; it < ITERS; ++it) {
    const int cur = it & 1, nxt = cur ^ 1;
    mpc_pgd_iter<<<dim3(16, 8), 256, 0, stream>>>(Hh, uhs[cur], ufs[cur], F,
                                                  ufs[nxt], uhs[nxt]);
  }
}
